// KalmanRefineNetV0_35888746726131
// MI455X (gfx1250) — matrix operand order repaired and checked by reading, not compiled, not run on hardware
//
#include <hip/hip_runtime.h>

// ---------------------------------------------------------------------------
// KalmanRefineNet scan for MI455X (gfx1250, wave32, WMMA).
//
//   B=64, L=64, C=256, H*W=64.
//   Fused per-timestep kernel: two bf16 WMMA GEMMs (gain + prediction,
//   M=256, N=4096, K=256) + sigmoid gate + Kalman blend in f32.
//   63 sequential launches provide the scan dependency; weights converted
//   to bf16 once; z-hat state ping-pongs through workspace.
// ---------------------------------------------------------------------------

typedef __bf16 v16bf __attribute__((ext_vector_type(16)));
typedef __bf16 v8bf  __attribute__((ext_vector_type(8)));
typedef float  v8f   __attribute__((ext_vector_type(8)));

#define C_DIM 256
#define L_DIM 64
#define HW_DIM 64

__device__ __forceinline__ unsigned short f32_to_bf16_rne(float f) {
  unsigned u = __builtin_bit_cast(unsigned, f);
  u += 0x7FFFu + ((u >> 16) & 1u);   // round-to-nearest-even
  return (unsigned short)(u >> 16);
}

__device__ __forceinline__ v16bf cat8(v8bf lo, v8bf hi) {
  return __builtin_shufflevector(lo, hi, 0, 1, 2, 3, 4, 5, 6, 7,
                                         8, 9, 10, 11, 12, 13, 14, 15);
}

// Convert both 256x256 weight matrices to bf16 (row-major [o][c]).
__global__ __launch_bounds__(256) void wconv_kernel(
    const float* __restrict__ Wg, const float* __restrict__ Wp,
    unsigned short* __restrict__ WgB, unsigned short* __restrict__ WpB) {
  int idx = blockIdx.x * 256 + threadIdx.x;    // 65536 total
  WgB[idx] = f32_to_bf16_rne(Wg[idx]);
  WpB[idx] = f32_to_bf16_rne(Wp[idx]);
}

// zhat0[b][c][hw] = z[b][0][c][hw]
__global__ __launch_bounds__(256) void init_kernel(
    const float* __restrict__ z, float* __restrict__ zhat) {
  int idx = blockIdx.x * 256 + threadIdx.x;    // 1,048,576 total
  int b = idx >> 14;                            // / (256*64)
  int rem = idx & 16383;
  zhat[idx] = z[((size_t)b << 20) + rem];       // l = 0 slice
}

// One scan step l:
//   g  = sigmoid(Wg @ z[:,l] + bg)
//   z' = Wp @ zhat_prev + bp
//   zhat_new = z' + g * (z[:,l] - z')
// Grid: 256 WGs (one per 16-position tile), 256 threads (8 wave32).
__global__ __launch_bounds__(256) void step_kernel(
    const float* __restrict__ z,
    const unsigned short* __restrict__ WgB,
    const unsigned short* __restrict__ WpB,
    const float* __restrict__ bg,
    const float* __restrict__ bp,
    const float* __restrict__ zprev,   // [b][c][hw]
    float* __restrict__ znext,         // [b][c][hw]
    int l) {
  constexpr int RS = 272;              // LDS row stride (bf16 elems), 544B rows
  __shared__ __align__(16) unsigned short zs[16 * RS];  // z tile  [n][c] bf16
  __shared__ __align__(16) unsigned short hs[16 * RS];  // ẑ tile [n][c] bf16

  const int tid  = threadIdx.x;
  const int tile = blockIdx.x;         // 0..255
  const int b    = tile >> 2;          // 4 tiles of 16 positions per batch
  const int hw0  = (tile & 3) << 4;

  const float* __restrict__ zb = z + (((size_t)b * L_DIM + l) * C_DIM) * HW_DIM;
  const float* __restrict__ hb = zprev + ((size_t)b * C_DIM) * HW_DIM;

  // Stage both B operands into LDS, transposed to [n][c], converted to bf16.
  for (int idx = tid; idx < 16 * C_DIM; idx += 256) {
    int c = idx >> 4;
    int n = idx & 15;
    zs[n * RS + c] = f32_to_bf16_rne(zb[c * HW_DIM + hw0 + n]);
    hs[n * RS + c] = f32_to_bf16_rne(hb[c * HW_DIM + hw0 + n]);
    if (l + 1 < L_DIM)  // warm L2 for the next timestep's z tile
      __builtin_prefetch(zb + (size_t)C_DIM * HW_DIM + c * HW_DIM + hw0 + n, 0, 1);
  }
  __syncthreads();

  const int wave = tid >> 5;
  const int lane = tid & 31;
  const int half = lane >> 4;          // ISA 16x32 A / 32x16 B lane halves
  const int lrow = lane & 15;
  const int kA = half << 3;            // A: K base 0 or 8 (plus +16 chunk)
  const int kB = half << 3;            // B: K base 0 or 8 (plus +16 chunk)

  v8f accg0 = {}, accg1 = {}, accp0 = {}, accp1 = {};

  const int m0 = (wave * 2) * 16;      // two 16-row M tiles per wave
  const int m1 = m0 + 16;
  const unsigned short* __restrict__ wg0 = WgB + (m0 + lrow) * C_DIM + kA;
  const unsigned short* __restrict__ wg1 = WgB + (m1 + lrow) * C_DIM + kA;
  const unsigned short* __restrict__ wp0 = WpB + (m0 + lrow) * C_DIM + kA;
  const unsigned short* __restrict__ wp1 = WpB + (m1 + lrow) * C_DIM + kA;
  const unsigned short* __restrict__ zr  = &zs[lrow * RS + kB];
  const unsigned short* __restrict__ hr  = &hs[lrow * RS + kB];

#pragma unroll
  for (int kk = 0; kk < C_DIM; kk += 32) {
    // B operands: one column n=lrow, K chunks [kB, kB+8) and [kB+16, kB+24).
    v16bf Bz = cat8(*(const v8bf*)(zr + kk), *(const v8bf*)(zr + kk + 16));
    v16bf Bh = cat8(*(const v8bf*)(hr + kk), *(const v8bf*)(hr + kk + 16));

    // A operands: row m+lrow, K chunks [kA, kA+8) and [kA+16, kA+24).
    v16bf Ag0 = cat8(*(const v8bf*)(wg0 + kk), *(const v8bf*)(wg0 + kk + 16));
    v16bf Ag1 = cat8(*(const v8bf*)(wg1 + kk), *(const v8bf*)(wg1 + kk + 16));
    v16bf Ap0 = cat8(*(const v8bf*)(wp0 + kk), *(const v8bf*)(wp0 + kk + 16));
    v16bf Ap1 = cat8(*(const v8bf*)(wp1 + kk), *(const v8bf*)(wp1 + kk + 16));

    accg0 = __builtin_amdgcn_wmma_f32_16x16x32_bf16(false, Ag0, false, Bz,
                                                    (short)0, accg0, false, false);
    accg1 = __builtin_amdgcn_wmma_f32_16x16x32_bf16(false, Ag1, false, Bz,
                                                    (short)0, accg1, false, false);
    accp0 = __builtin_amdgcn_wmma_f32_16x16x32_bf16(false, Ap0, false, Bh,
                                                    (short)0, accp0, false, false);
    accp1 = __builtin_amdgcn_wmma_f32_16x16x32_bf16(false, Ap1, false, Bh,
                                                    (short)0, accp1, false, false);
  }

  // Epilogue: C/D layout => element r, lane -> (o = m + r + 8*half, n = lrow).
  const int ncol = hw0 + lrow;
  float* __restrict__ ob = znext + ((size_t)b * C_DIM) * HW_DIM;

#pragma unroll
  for (int r = 0; r < 8; ++r) {
    {
      int o = m0 + r + (half << 3);
      float gate = 1.0f / (1.0f + __expf(-(accg0[r] + bg[o])));
      float zp = accp0[r] + bp[o];
      float zi = zb[o * HW_DIM + ncol];          // exact f32 z_i for the blend
      ob[o * HW_DIM + ncol] = zp + gate * (zi - zp);
    }
    {
      int o = m1 + r + (half << 3);
      float gate = 1.0f / (1.0f + __expf(-(accg1[r] + bg[o])));
      float zp = accp1[r] + bp[o];
      float zi = zb[o * HW_DIM + ncol];
      ob[o * HW_DIM + ncol] = zp + gate * (zi - zp);
    }
  }
}

extern "C" void kernel_launch(void* const* d_in, const int* in_sizes, int n_in,
                              void* d_out, int out_size, void* d_ws, size_t ws_size,
                              hipStream_t stream) {
  const float* z  = (const float*)d_in[0];   // (64,64,256,8,8)
  const float* Wg = (const float*)d_in[1];   // (256,256)
  const float* bg = (const float*)d_in[2];   // (256,)
  const float* Wp = (const float*)d_in[3];   // (256,256)
  const float* bp = (const float*)d_in[4];   // (256,)
  float* out = (float*)d_out;                // (64,256,8,8)

  // Workspace layout (~8.6 MB):
  //   [0, 128K)        W_gain bf16
  //   [128K, 256K)     W_pred bf16
  //   [256K, 256K+4M)  zhat buffer 0
  //   [.., +4M)        zhat buffer 1
  char* ws = (char*)d_ws;
  unsigned short* WgB = (unsigned short*)ws;
  unsigned short* WpB = (unsigned short*)(ws + (size_t)131072);
  float* buf0 = (float*)(ws + (size_t)262144);
  float* buf1 = (float*)(ws + (size_t)262144 + (size_t)4 * 1024 * 1024);
  float* bufs[2] = {buf0, buf1};

  wconv_kernel<<<256, 256, 0, stream>>>(Wg, Wp, WgB, WpB);
  init_kernel<<<4096, 256, 0, stream>>>(z, buf0);

  for (int l = 1; l < L_DIM; ++l) {
    const float* prev = bufs[(l - 1) & 1];
    float* next = (l == L_DIM - 1) ? out : bufs[l & 1];
    step_kernel<<<256, 256, 0, stream>>>(z, WgB, WpB, bg, bp, prev, next, l);
  }
}
